// PartitionAttentionLayer_48163763258058
// MI455X (gfx1250) — matrix operand order repaired and checked by reading, not compiled, not run on hardware
//
#include <hip/hip_runtime.h>
#include <hip/hip_bf16.h>
#include <math.h>

// ---------------- problem constants ----------------
#define BATCH 16
#define CH    256
#define HH    56
#define WW    56
#define PH    7
#define PW    7
#define PTOK  49          // tokens per window
#define PPAD  64          // padded tokens (4 x 16 tiles)
#define GWIN  64          // windows per image
#define HEADS 8
#define DH    32
#define NTOK  (BATCH * GWIN * PTOK)   // 50176 = 3136 * 16
#define MTILES (NTOK / 16)            // 3136
#define MLPC  1024
#define SCALE_K 0.0625f               // C^-0.5 = 1/16

typedef __bf16 bf16_t;
typedef bf16_t v16bf __attribute__((ext_vector_type(16)));
typedef bf16_t v8bf  __attribute__((ext_vector_type(8)));
typedef float  v8f   __attribute__((ext_vector_type(8)));

// ---------------- WMMA fragment helpers (wave32) ----------------
// A (16x32 bf16): lane l holds row m=l&15; K = ko..ko+15 contiguous, ko = (l>=16)*16
__device__ __forceinline__ v16bf load_a_frag(const bf16_t* __restrict__ base, int ld) {
    const int l  = threadIdx.x & 31;
    const int m  = l & 15;
    const int ko = (l >> 4) << 4;               // 0 or 16
    const bf16_t* p = base + (size_t)m * ld + ko;
    v8bf lo = *(const v8bf*)(p);
    v8bf hi = *(const v8bf*)(p + 8);
    v16bf r;
#pragma unroll
    for (int i = 0; i < 8; ++i) { r[i] = lo[i]; r[i + 8] = hi[i]; }
    return r;
}

// B (32x16 bf16) from Wt stored [N,K] row-major: lane l holds col n=l&15;
// K = ko..ko+15 contiguous, ko = (l>=16)*16  -> one 32B load
__device__ __forceinline__ v16bf load_b_frag(const bf16_t* __restrict__ base, int ld) {
    const int l  = threadIdx.x & 31;
    const int n  = l & 15;
    const int ko = (l >> 4) << 4;               // 0 or 16
    return *(const v16bf*)(base + (size_t)n * ld + ko);
}

__device__ __forceinline__ v8f wmma_bf16(v16bf a, v16bf b, v8f c) {
    return __builtin_amdgcn_wmma_f32_16x16x32_bf16(false, a, false, b, (short)0, c,
                                                   false, false);
}

// ---------------- weight convert + transpose: fp32 [K,N] -> bf16 [N,K] ----------------
__global__ void convert_transpose_kernel(const float* __restrict__ in,
                                         bf16_t* __restrict__ out, int K, int N) {
    int idx = blockIdx.x * 256 + threadIdx.x;
    if (idx < K * N) {
        int n = idx / K, k = idx - n * K;
        out[idx] = (bf16_t)in[(size_t)k * N + n];
    }
}

// ---------------- expand relative-position bias into padded [HEADS,64,64] fp32 ----------
// cols >= 49 carry -1e30 (softmax mask); rows >= 49 carry 0 (never stored downstream)
__global__ void expand_bias_kernel(const float* __restrict__ bias_table,
                                   const int* __restrict__ rel_idx,
                                   float* __restrict__ bexp) {
    int idx = blockIdx.x * 256 + threadIdx.x;   // HEADS*64*64 = 32768
    if (idx < HEADS * PPAD * PPAD) {
        int h = idx >> 12;
        int r = idx & 4095;
        int i = r >> 6, j = r & 63;
        float v;
        if (j >= PTOK)      v = -1e30f;
        else if (i >= PTOK) v = 0.f;
        else                v = bias_table[rel_idx[i * PTOK + j] * HEADS + h];
        bexp[idx] = v;
    }
}

// ---------------- LN1 + window partition: x[B,C,H,W] -> y bf16 [NTOK, C] ----------------
__global__ __launch_bounds__(256) void ln1_partition_kernel(
    const float* __restrict__ x, const float* __restrict__ gamma,
    const float* __restrict__ beta, bf16_t* __restrict__ y) {
    __shared__ float tile[CH * 51];             // pad 49->51: conflict-free (gcd(51,64)=1)
    const int bg = blockIdx.x;
    const int b  = bg >> 6, g = bg & 63;
    const int gh = g >> 3, gw = g & 7;
    const int t  = threadIdx.x;

    for (int idx = t; idx < CH * PTOK; idx += 256) {
        int c = idx / PTOK, p = idx - c * PTOK;
        int hh = gh * PH + p / PW, ww = gw * PW + p % PW;
        tile[c * 51 + p] = x[(((size_t)b * CH + c) * HH + hh) * WW + ww];
    }
    __syncthreads();

    const int wave = t >> 5, l = t & 31;
    for (int p = wave; p < PTOK; p += 8) {
        float s = 0.f, s2 = 0.f;
#pragma unroll
        for (int i = 0; i < 8; ++i) {
            float v = tile[(l + 32 * i) * 51 + p];
            s += v; s2 += v * v;
        }
#pragma unroll
        for (int m = 16; m >= 1; m >>= 1) {
            s  += __shfl_xor(s,  m, 32);
            s2 += __shfl_xor(s2, m, 32);
        }
        float mu  = s * (1.f / CH);
        float var = s2 * (1.f / CH) - mu * mu;
        float rs  = rsqrtf(var + 1e-5f);
        size_t row = ((size_t)bg * PTOK + p) * CH;
#pragma unroll
        for (int i = 0; i < 8; ++i) {
            int c = l + 32 * i;
            float v = (tile[c * 51 + p] - mu) * rs * gamma[c] + beta[c];
            y[row + c] = (bf16_t)v;
        }
    }
}

// ---------------- LN2: x1 fp32 [NTOK,C] -> z bf16 [NTOK,C] ----------------
__global__ __launch_bounds__(256) void ln2_kernel(
    const float* __restrict__ x1, const float* __restrict__ gamma,
    const float* __restrict__ beta, bf16_t* __restrict__ z) {
    const int tok = blockIdx.x * 8 + (threadIdx.x >> 5);
    const int l   = threadIdx.x & 31;
    const float* row = x1 + (size_t)tok * CH;
    float vals[8], s = 0.f, s2 = 0.f;
#pragma unroll
    for (int i = 0; i < 8; ++i) {
        float v = row[l + 32 * i];
        vals[i] = v; s += v; s2 += v * v;
    }
#pragma unroll
    for (int m = 16; m >= 1; m >>= 1) {
        s  += __shfl_xor(s,  m, 32);
        s2 += __shfl_xor(s2, m, 32);
    }
    float mu  = s * (1.f / CH);
    float var = s2 * (1.f / CH) - mu * mu;
    float rs  = rsqrtf(var + 1e-5f);
    bf16_t* zr = z + (size_t)tok * CH;
#pragma unroll
    for (int i = 0; i < 8; ++i) {
        int c = l + 32 * i;
        zr[c] = (bf16_t)((vals[i] - mu) * rs * gamma[c] + beta[c]);
    }
}

// ---------------- generic WMMA GEMM ----------------
// Each wave computes a 32x64 output strip (2 M-tiles x 4 N-tiles):
// per K-step: 2 A-frag + 4 B-frag loads feed 8 WMMAs (B reused 2x, 8 indep accums).
struct EpiArgs {
    bf16_t* q; bf16_t* k; bf16_t* v;            // EPI 0
    const float* xin; float* x1;                // EPI 1
    bf16_t* h;                                  // EPI 2
    const float* x1res; float* out;             // EPI 3
};

template <int EPI>
__global__ __launch_bounds__(128) void gemm_bf16_kernel(
    const bf16_t* __restrict__ A, const bf16_t* __restrict__ Wt,
    const float* __restrict__ bias, int K, EpiArgs e) {
    const int mbase = blockIdx.x * 32;          // 2 M-tiles per wave
    const int wave  = threadIdx.x >> 5;
    const int nbase = (blockIdx.y * 4 + wave) * 64;

    v8f acc[2][4];
#pragma unroll
    for (int mt = 0; mt < 2; ++mt)
#pragma unroll
        for (int t = 0; t < 4; ++t) acc[mt][t] = v8f{};

    const bf16_t* Arow0 = A + (size_t)mbase * K;
    const bf16_t* Arow1 = Arow0 + (size_t)16 * K;
    for (int kb = 0; kb < K; kb += 32) {
        v16bf a0 = load_a_frag(Arow0 + kb, K);
        v16bf a1 = load_a_frag(Arow1 + kb, K);
#pragma unroll
        for (int t = 0; t < 4; ++t) {
            v16bf b = load_b_frag(Wt + (size_t)(nbase + t * 16) * K + kb, K);
            acc[0][t] = wmma_bf16(a0, b, acc[0][t]);
            acc[1][t] = wmma_bf16(a1, b, acc[1][t]);
        }
    }

    const int l    = threadIdx.x & 31;
    const int half = l >> 4;
    const int lc   = l & 15;
#pragma unroll
    for (int mt = 0; mt < 2; ++mt) {
#pragma unroll
        for (int t = 0; t < 4; ++t) {
            const int n  = nbase + t * 16 + lc;
            const float bn = bias[n];
#pragma unroll
            for (int j = 0; j < 8; ++j) {
                const int m = mbase + mt * 16 + j + 8 * half;
                float v = acc[mt][t][j] + bn;
                if constexpr (EPI == 0) {            // QKV scatter (padded, k pre-scaled)
                    int bg = m / PTOK, p = m - bg * PTOK;
                    int which = n >> 8;              // 0=q 1=k 2=v
                    int c = n & 255;
                    int hd = c >> 5, d = c & 31;
                    size_t bh = (size_t)(bg * HEADS + hd);
                    if (which == 0)      e.q[(bh * PPAD + p) * DH + d] = (bf16_t)v;
                    else if (which == 1) e.k[(bh * PPAD + p) * DH + d] = (bf16_t)(v * SCALE_K);
                    else                 e.v[(bh * DH + d) * PPAD + p] = (bf16_t)v;
                } else if constexpr (EPI == 1) {     // merge + residual gather -> x1 fp32
                    int bg = m / PTOK, p = m - bg * PTOK;
                    int b = bg >> 6, g = bg & 63;
                    int gh = g >> 3, gw = g & 7;
                    int hh = gh * PH + p / PW, ww = gw * PW + p % PW;
                    float xv = e.xin[(((size_t)b * CH + n) * HH + hh) * WW + ww];
                    e.x1[(size_t)m * CH + n] = v + xv;
                } else if constexpr (EPI == 2) {     // exact GELU -> h bf16
                    float gl = 0.5f * v * (1.f + erff(v * 0.70710678118f));
                    e.h[(size_t)m * MLPC + n] = (bf16_t)gl;
                } else {                             // residual + departition -> d_out fp32
                    int bg = m / PTOK, p = m - bg * PTOK;
                    int b = bg >> 6, g = bg & 63;
                    int gh = g >> 3, gw = g & 7;
                    int hh = gh * PH + p / PW, ww = gw * PW + p % PW;
                    float r = e.x1res[(size_t)m * CH + n];
                    e.out[(((size_t)b * CH + n) * HH + hh) * WW + ww] = v + r;
                }
            }
        }
    }
}

// ---------------- windowed attention: one block per (bg, head) ----------------
__global__ __launch_bounds__(128) void attn_kernel(
    const bf16_t* __restrict__ q_ws, const bf16_t* __restrict__ k_ws,
    const bf16_t* __restrict__ v_ws, const float* __restrict__ bexp,
    bf16_t* __restrict__ attn_out) {
    __shared__ __align__(128) bf16_t probs[4][16][PPAD];   // 8 KB
    const int bgh  = blockIdx.x;
    const int h    = bgh & 7;
    const int bg   = bgh >> 3;
    const int wave = threadIdx.x >> 5;
    const int l    = threadIdx.x & 31;
    const int half = l >> 4;
    const int lc   = l & 15;

    const bf16_t* qb = q_ws + (size_t)bgh * PPAD * DH;
    const bf16_t* kb = k_ws + (size_t)bgh * PPAD * DH;
    const bf16_t* vb = v_ws + (size_t)bgh * DH * PPAD;
    const float*  bh = bexp + (size_t)h * PPAD * PPAD;

    // S = Q @ K^T  (K already scaled); one k-step since DH == 32
    v16bf qf = load_a_frag(qb + (size_t)wave * 16 * DH, DH);
    v8f s[4];
#pragma unroll
    for (int nt = 0; nt < 4; ++nt) {
        v16bf kf = load_b_frag(kb + (size_t)nt * 16 * DH, DH);
        v8f z = {};
        s[nt] = wmma_bf16(qf, kf, z);
    }

    // + expanded bias (mask for cols>=49 baked in): 32 branchless coalesced loads
#pragma unroll
    for (int nt = 0; nt < 4; ++nt) {
        int jj = nt * 16 + lc;
#pragma unroll
        for (int j = 0; j < 8; ++j) {
            int i = wave * 16 + j + 8 * half;
            s[nt][j] += bh[i * PPAD + jj];
        }
    }

    // row softmax: a row lives in one accumulator slot across 16 lanes of a half-wave
#pragma unroll
    for (int j = 0; j < 8; ++j) {
        float mx = fmaxf(fmaxf(s[0][j], s[1][j]), fmaxf(s[2][j], s[3][j]));
#pragma unroll
        for (int mk = 8; mk >= 1; mk >>= 1) mx = fmaxf(mx, __shfl_xor(mx, mk, 32));
        float sum = 0.f;
#pragma unroll
        for (int nt = 0; nt < 4; ++nt) {
            float ev = __expf(s[nt][j] - mx);
            s[nt][j] = ev; sum += ev;
        }
#pragma unroll
        for (int mk = 8; mk >= 1; mk >>= 1) sum += __shfl_xor(sum, mk, 32);
        float inv = 1.f / sum;
#pragma unroll
        for (int nt = 0; nt < 4; ++nt) s[nt][j] *= inv;
    }

    // probs D-layout -> LDS -> A-layout
#pragma unroll
    for (int nt = 0; nt < 4; ++nt) {
        int jj = nt * 16 + lc;
#pragma unroll
        for (int j = 0; j < 8; ++j)
            probs[wave][j + 8 * half][jj] = (bf16_t)s[nt][j];
    }
    __syncthreads();

    // O = P[16x64] @ V[64x32]
#pragma unroll
    for (int dt = 0; dt < 2; ++dt) {
        v8f o = {};
#pragma unroll
        for (int ks = 0; ks < 2; ++ks) {
            v16bf a = load_a_frag(&probs[wave][0][ks * 32], PPAD);
            v16bf b = load_b_frag(vb + (size_t)(dt * 16) * PPAD + ks * 32, PPAD);
            o = wmma_bf16(a, b, o);
        }
#pragma unroll
        for (int j = 0; j < 8; ++j) {
            int i = wave * 16 + j + 8 * half;
            if (i < PTOK) {
                int d = dt * 16 + lc;
                attn_out[((size_t)bg * PTOK + i) * CH + h * DH + d] = (bf16_t)o[j];
            }
        }
    }
}

// ---------------- host-side launch ----------------
extern "C" void kernel_launch(void* const* d_in, const int* in_sizes, int n_in,
                              void* d_out, int out_size, void* d_ws, size_t ws_size,
                              hipStream_t stream) {
    (void)in_sizes; (void)n_in; (void)out_size; (void)ws_size;
    const float* x          = (const float*)d_in[0];
    const float* ln1_g      = (const float*)d_in[1];
    const float* ln1_b      = (const float*)d_in[2];
    const float* wqkv       = (const float*)d_in[3];
    const float* bqkv       = (const float*)d_in[4];
    const float* wmerge     = (const float*)d_in[5];
    const float* bmerge     = (const float*)d_in[6];
    const float* bias_table = (const float*)d_in[7];
    const float* ln2_g      = (const float*)d_in[8];
    const float* ln2_b      = (const float*)d_in[9];
    const float* w1         = (const float*)d_in[10];
    const float* b1         = (const float*)d_in[11];
    const float* w2         = (const float*)d_in[12];
    const float* b2         = (const float*)d_in[13];
    const int*   rel_idx    = (const int*)d_in[14];
    float* out = (float*)d_out;

    char* ws = (char*)d_ws;
    auto alloc = [&](size_t bytes) {
        void* p = ws;
        ws += (bytes + 255) & ~(size_t)255;
        return p;
    };
    bf16_t* wqkvT   = (bf16_t*)alloc((size_t)768 * CH * 2);
    bf16_t* wmergeT = (bf16_t*)alloc((size_t)CH * CH * 2);
    bf16_t* w1T     = (bf16_t*)alloc((size_t)MLPC * CH * 2);
    bf16_t* w2T     = (bf16_t*)alloc((size_t)CH * MLPC * 2);
    float*  bexp    = (float*)alloc((size_t)HEADS * PPAD * PPAD * 4);
    bf16_t* actA    = (bf16_t*)alloc((size_t)NTOK * CH * 2);   // y -> attn_out -> z (reused)
    const size_t qkv_one = (size_t)BATCH * GWIN * HEADS * PPAD * DH * 2;
    bf16_t* q_ws    = (bf16_t*)alloc(qkv_one);
    bf16_t* k_ws    = (bf16_t*)alloc(qkv_one);
    bf16_t* v_ws    = (bf16_t*)alloc(qkv_one);
    float*  x1      = (float*)alloc((size_t)NTOK * CH * 4);
    bf16_t* h_ws    = (bf16_t*)alloc((size_t)NTOK * MLPC * 2);

    // 1) convert + transpose weights to bf16 [N,K]; expand bias table
    convert_transpose_kernel<<<(CH * 768 + 255) / 256, 256, 0, stream>>>(wqkv, wqkvT, CH, 768);
    convert_transpose_kernel<<<(CH * CH + 255) / 256, 256, 0, stream>>>(wmerge, wmergeT, CH, CH);
    convert_transpose_kernel<<<(CH * MLPC + 255) / 256, 256, 0, stream>>>(w1, w1T, CH, MLPC);
    convert_transpose_kernel<<<(MLPC * CH + 255) / 256, 256, 0, stream>>>(w2, w2T, MLPC, CH);
    expand_bias_kernel<<<(HEADS * PPAD * PPAD + 255) / 256, 256, 0, stream>>>(bias_table,
                                                                             rel_idx, bexp);

    // 2) zero padded q/k/v (rows 49..63 must be zero)
    hipMemsetAsync(q_ws, 0, 3 * qkv_one, stream);

    // 3) LN1 + window partition -> y bf16
    ln1_partition_kernel<<<BATCH * GWIN, 256, 0, stream>>>(x, ln1_g, ln1_b, actA);

    EpiArgs e{};
    // 4) QKV GEMM: [NTOK,256] x [256,768]
    e.q = q_ws; e.k = k_ws; e.v = v_ws;
    gemm_bf16_kernel<0><<<dim3(MTILES / 2, 3), 128, 0, stream>>>(actA, wqkvT, bqkv, CH, e);

    // 5) windowed attention (actA is reused as attn_out; y is dead)
    attn_kernel<<<BATCH * GWIN * HEADS, 128, 0, stream>>>(q_ws, k_ws, v_ws, bexp, actA);

    // 6) merge GEMM + residual gather -> x1 fp32
    e = EpiArgs{}; e.xin = x; e.x1 = x1;
    gemm_bf16_kernel<1><<<dim3(MTILES / 2, 1), 128, 0, stream>>>(actA, wmergeT, bmerge, CH, e);

    // 7) LN2 -> z bf16 (reuse actA)
    ln2_kernel<<<NTOK / 8, 256, 0, stream>>>(x1, ln2_g, ln2_b, actA);

    // 8) MLP fc1 + exact GELU -> h bf16
    e = EpiArgs{}; e.h = h_ws;
    gemm_bf16_kernel<2><<<dim3(MTILES / 2, 4), 128, 0, stream>>>(actA, w1T, b1, CH, e);

    // 9) MLP fc2 + residual + departition -> d_out fp32
    e = EpiArgs{}; e.x1res = x1; e.out = out;
    gemm_bf16_kernel<3><<<dim3(MTILES / 2, 1), 128, 0, stream>>>(h_ws, w2T, b2, MLPC, e);
}
